// BiMambaBlock_7894149890252
// MI455X (gfx1250) — matrix operand order repaired and checked by reading, not compiled, not run on hardware
//
#include <hip/hip_runtime.h>
#include <hip/hip_bf16.h>
#include <math.h>

// ---------------------------------------------------------------------------
// BiMamba block forward for MI455X (gfx1250), wave32 + WMMA f32_16x16x32_f16
// Register-blocked GEMM: each wave owns a 32x64 macro-tile (2x4 WMMA tiles).
// ---------------------------------------------------------------------------

#define B_SZ     4
#define SEQ      1024
#define D_MODEL  1024
#define D_INNER  2048
#define D_STATE  16
#define D_CONV   4
#define DT_RANK  64
#define XPROJ_N  (DT_RANK + 2 * D_STATE)   // 96
#define ROWS     (B_SZ * SEQ)              // 4096 (GEMM M)

typedef _Float16 f16;
typedef _Float16 v16h __attribute__((ext_vector_type(16)));
typedef _Float16 v8h  __attribute__((ext_vector_type(8)));
typedef float    v8f  __attribute__((ext_vector_type(8)));

// ------------------------- generic fp32 -> fp16 convert ---------------------
__global__ __launch_bounds__(256) void k_f32_to_f16(const float* __restrict__ in,
                                                    f16* __restrict__ out, int n) {
  for (int i = blockIdx.x * 256 + threadIdx.x; i < n; i += gridDim.x * 256)
    out[i] = (f16)in[i];
}

// extract dt columns (first 64 of 96) from dbl into a packed f16 matrix
__global__ __launch_bounds__(256) void k_extract_dt(const float* __restrict__ dbl,
                                                    f16* __restrict__ dt16, int n) {
  for (int i = blockIdx.x * 256 + threadIdx.x; i < n; i += gridDim.x * 256) {
    int row = i / DT_RANK, c = i % DT_RANK;
    dt16[i] = (f16)dbl[row * XPROJ_N + c];
  }
}

// ------------------------------ LayerNorm -----------------------------------
__global__ __launch_bounds__(256) void k_layernorm(const float* __restrict__ x,
                                                   const float* __restrict__ g,
                                                   const float* __restrict__ be,
                                                   f16* __restrict__ xn,
                                                   f16* __restrict__ xnr) {
  __shared__ float s1[256], s2[256];
  const int row = blockIdx.x;
  const int tid = threadIdx.x;
  const float* xr = x + (size_t)row * D_MODEL;
  float a = 0.f, b2 = 0.f;
  for (int c = tid; c < D_MODEL; c += 256) { float v = xr[c]; a += v; b2 += v * v; }
  s1[tid] = a; s2[tid] = b2; __syncthreads();
  for (int st = 128; st > 0; st >>= 1) {
    if (tid < st) { s1[tid] += s1[tid + st]; s2[tid] += s2[tid + st]; }
    __syncthreads();
  }
  const float mean = s1[0] * (1.f / D_MODEL);
  const float var  = s2[0] * (1.f / D_MODEL) - mean * mean;
  const float inv  = rsqrtf(var + 1e-5f);
  const int b = row >> 10, l = row & (SEQ - 1);
  const size_t ro = (size_t)row * D_MODEL;
  const size_t rr = ((size_t)(b << 10) + (SEQ - 1 - l)) * D_MODEL;
  for (int c = tid; c < D_MODEL; c += 256) {
    float v = (xr[c] - mean) * inv * g[c] + be[c];
    xn[ro + c]  = (f16)v;
    xnr[rr + c] = (f16)v;
  }
}

// ---------------------- epilogue helper (shared logic) ----------------------
// modes: 0 f32 store | 1 softplus(acc+bias) f32 | 2 f16 store (+flip,+colOff)
//        3 acc+bias+resid f32
__device__ __forceinline__ void epilogue_store(
    float v, int m, int ncol, int N,
    float* __restrict__ outF, f16* __restrict__ outH, int ldo,
    const float* __restrict__ bias, const float* __restrict__ resid,
    int mode, int colOff, int flipL) {
  if (mode == 0) {
    outF[(size_t)m * ldo + ncol] = v;
  } else if (mode == 1) {
    v += bias[ncol];
    v = (v > 20.f) ? v : log1pf(__expf(v));
    outF[(size_t)m * ldo + ncol] = v;
  } else if (mode == 2) {
    int mb = m >> 10, ml = m & (SEQ - 1);
    int m2 = flipL ? ((mb << 10) + (SEQ - 1 - ml)) : m;
    outH[(size_t)m2 * ldo + colOff + ncol] = (f16)v;
  } else {
    v += bias[ncol] + resid[(size_t)m * N + ncol];
    outF[(size_t)m * ldo + ncol] = v;
  }
}

// ------------------- simple WMMA GEMM (one 16x16 tile/wave) -----------------
// used only for small-N GEMMs (x_proj, N=96)
__global__ __launch_bounds__(256)
void k_wmma_gemm(const f16* __restrict__ A, int lda,
                 const f16* __restrict__ W, int ldw,
                 int M, int N, int K,
                 float* __restrict__ outF, f16* __restrict__ outH, int ldo,
                 const float* __restrict__ bias, const float* __restrict__ resid,
                 int mode, int colOff, int flipL) {
  const int lane = threadIdx.x & 31;
  const int wave = threadIdx.x >> 5;
  const int tilesN = N >> 4;
  const int tilesM = M >> 4;
  const int tile = blockIdx.x * 8 + wave;
  if (tile >= tilesM * tilesN) return;          // wave-uniform
  const int tm = tile / tilesN;
  const int tn = tile - tm * tilesN;
  const int g = lane >> 4;
  const int r = lane & 15;
  const f16* Ap = A + (size_t)(tm * 16 + r) * lda;
  const f16* Wp = W + (size_t)(tn * 16 + r) * ldw;
  v8f acc = {0.f, 0.f, 0.f, 0.f, 0.f, 0.f, 0.f, 0.f};
  for (int kb = 0; kb < K; kb += 32) {
    v16h a = *(const v16h*)(Ap + kb + 16 * g);
    v16h b = *(const v16h*)(Wp + kb + 16 * g);
    acc = __builtin_amdgcn_wmma_f32_16x16x32_f16(
        false, a, false, b, (short)0, acc, false, false);
  }
  const int ncol = tn * 16 + r;
#pragma unroll
  for (int e = 0; e < 8; ++e)
    epilogue_store(acc[e], tm * 16 + 8 * g + e, ncol, N,
                   outF, outH, ldo, bias, resid, mode, colOff, flipL);
}

// -------------- register-blocked WMMA GEMM (32x64 macro-tile) ---------------
// requires M%32==0, N%64==0, K%32==0. 8 WMMAs per 32-deep K-step; A fragments
// reused across 4 N-tiles, B fragments across 2 M-tiles.
__global__ __launch_bounds__(256)
void k_wmma_gemm_blk(const f16* __restrict__ A, int lda,
                     const f16* __restrict__ W, int ldw,
                     int M, int N, int K,
                     float* __restrict__ outF, f16* __restrict__ outH, int ldo,
                     const float* __restrict__ bias, const float* __restrict__ resid,
                     int mode, int colOff, int flipL) {
  const int lane = threadIdx.x & 31;
  const int wave = threadIdx.x >> 5;
  const int mtN = N >> 6;                       // macro-tiles along N (64 wide)
  const int mtM = M >> 5;                       // macro-tiles along M (32 tall)
  const int tile = blockIdx.x * 8 + wave;
  if (tile >= mtM * mtN) return;                // wave-uniform
  const int tm = tile / mtN;
  const int tn = tile - tm * mtN;
  const int g = lane >> 4;
  const int r = lane & 15;

  const f16* Ap0 = A + (size_t)(tm * 32 + r) * lda;
  const f16* Ap1 = A + (size_t)(tm * 32 + 16 + r) * lda;
  const f16* Wp0 = W + (size_t)(tn * 64 + r) * ldw;
  const f16* Wp1 = W + (size_t)(tn * 64 + 16 + r) * ldw;
  const f16* Wp2 = W + (size_t)(tn * 64 + 32 + r) * ldw;
  const f16* Wp3 = W + (size_t)(tn * 64 + 48 + r) * ldw;

  v8f acc[2][4];
#pragma unroll
  for (int i = 0; i < 2; ++i)
#pragma unroll
    for (int j = 0; j < 4; ++j)
      acc[i][j] = (v8f){0.f, 0.f, 0.f, 0.f, 0.f, 0.f, 0.f, 0.f};

  for (int kb = 0; kb < K; kb += 32) {
    // A fragments (elems 0..15 <- K=kb+16g..kb+16g+15)
    v16h a[2];
    a[0] = *(const v16h*)(Ap0 + kb + 16 * g);
    a[1] = *(const v16h*)(Ap1 + kb + 16 * g);
    // B fragments (elems 0..15 <- K=kb+16g..kb+16g+15 of B column = W row)
    v16h b0 = *(const v16h*)(Wp0 + kb + 16 * g);
    v16h b1 = *(const v16h*)(Wp1 + kb + 16 * g);
    v16h b2 = *(const v16h*)(Wp2 + kb + 16 * g);
    v16h b3 = *(const v16h*)(Wp3 + kb + 16 * g);
    if (kb + 32 < K) {                          // stream next K-slice into L2
      __builtin_prefetch(Ap0 + kb + 32, 0, 1);
      __builtin_prefetch(Wp0 + kb + 32, 0, 1);
    }
#pragma unroll
    for (int i = 0; i < 2; ++i) {
      acc[i][0] = __builtin_amdgcn_wmma_f32_16x16x32_f16(
          false, a[i], false, b0, (short)0, acc[i][0], false, false);
      acc[i][1] = __builtin_amdgcn_wmma_f32_16x16x32_f16(
          false, a[i], false, b1, (short)0, acc[i][1], false, false);
      acc[i][2] = __builtin_amdgcn_wmma_f32_16x16x32_f16(
          false, a[i], false, b2, (short)0, acc[i][2], false, false);
      acc[i][3] = __builtin_amdgcn_wmma_f32_16x16x32_f16(
          false, a[i], false, b3, (short)0, acc[i][3], false, false);
    }
  }
  // D layout: lane (g, n=r), element e -> (M = 8g+e, N = n)
#pragma unroll
  for (int j = 0; j < 4; ++j) {
    const int ncol = tn * 64 + j * 16 + r;
#pragma unroll
    for (int i = 0; i < 2; ++i)
#pragma unroll
      for (int e = 0; e < 8; ++e)
        epilogue_store(acc[i][j][e], tm * 32 + i * 16 + 8 * g + e, ncol, N,
                       outF, outH, ldo, bias, resid, mode, colOff, flipL);
  }
}

// ---------------------- depthwise causal conv + SiLU ------------------------
__global__ __launch_bounds__(256) void k_conv_silu(const float* __restrict__ xz,
                                                   const float* __restrict__ cw,
                                                   const float* __restrict__ cb,
                                                   float* __restrict__ xc32,
                                                   f16* __restrict__ xc16) {
  const int total = ROWS * D_INNER;
  for (int idx = blockIdx.x * 256 + threadIdx.x; idx < total; idx += gridDim.x * 256) {
    const int d  = idx & (D_INNER - 1);
    const int bl = idx >> 11;                   // b*SEQ + l
    const int l  = bl & (SEQ - 1);
    const int b  = bl >> 10;
    float s = cb[d];
#pragma unroll
    for (int k = 0; k < D_CONV; ++k) {
      int ls = l - (D_CONV - 1) + k;
      if (ls >= 0)
        s += cw[d * D_CONV + k] *
             xz[((size_t)(b << 10) + ls) * (2 * D_INNER) + d];
    }
    float y = s / (1.f + __expf(-s));           // SiLU
    xc32[idx] = y;
    xc16[idx] = (f16)y;
  }
}

// --------------------------- selective scan ---------------------------------
__global__ __launch_bounds__(256) void k_scan(const float* __restrict__ delta,
                                              const float* __restrict__ u,
                                              const float* __restrict__ dbl,
                                              const float* __restrict__ xz,
                                              const float* __restrict__ A_log,
                                              const float* __restrict__ Dp,
                                              f16* __restrict__ yg16) {
  __shared__ float sBC[2 * D_STATE];            // B_t | C_t for this (b, l)
  const int b = blockIdx.x >> 3;
  const int d = ((blockIdx.x & 7) << 8) + threadIdx.x;
  float a[D_STATE], h[D_STATE];
#pragma unroll
  for (int s = 0; s < D_STATE; ++s) {
    a[s] = -__expf(A_log[d * D_STATE + s]);
    h[s] = 0.f;
  }
  const float Dv = Dp[d];
  for (int l = 0; l < SEQ; ++l) {
    const size_t bl = (size_t)(b << 10) + l;
    __syncthreads();
    if (threadIdx.x < 2 * D_STATE)
      sBC[threadIdx.x] = dbl[bl * XPROJ_N + DT_RANK + threadIdx.x];
    __syncthreads();
    const float dt = delta[bl * D_INNER + d];
    const float ut = u[bl * D_INNER + d];
    const float du = dt * ut;
    float y = 0.f;
#pragma unroll
    for (int s = 0; s < D_STATE; ++s) {
      h[s] = h[s] * __expf(dt * a[s]) + du * sBC[s];
      y += h[s] * sBC[D_STATE + s];
    }
    const float z = xz[bl * (2 * D_INNER) + D_INNER + d];
    const float gate = z / (1.f + __expf(-z));
    yg16[bl * D_INNER + d] = (f16)((y + ut * Dv) * gate);
  }
}

// ---------------------------------------------------------------------------
static inline void gemm(hipStream_t st, const f16* A, int lda, const f16* W, int ldw,
                        int M, int N, int K, float* outF, f16* outH, int ldo,
                        const float* bias, const float* resid,
                        int mode, int colOff, int flipL) {
  if ((M % 32 == 0) && (N % 64 == 0)) {
    int tiles = (M / 32) * (N / 64);
    int blocks = (tiles + 7) / 8;
    k_wmma_gemm_blk<<<blocks, 256, 0, st>>>(A, lda, W, ldw, M, N, K,
                                            outF, outH, ldo, bias, resid,
                                            mode, colOff, flipL);
  } else {
    int tiles = (M / 16) * (N / 16);
    int blocks = (tiles + 7) / 8;
    k_wmma_gemm<<<blocks, 256, 0, st>>>(A, lda, W, ldw, M, N, K,
                                        outF, outH, ldo, bias, resid,
                                        mode, colOff, flipL);
  }
}

extern "C" void kernel_launch(void* const* d_in, const int* in_sizes, int n_in,
                              void* d_out, int out_size, void* d_ws, size_t ws_size,
                              hipStream_t stream) {
  (void)in_sizes; (void)n_in; (void)out_size; (void)ws_size;
  const float* x     = (const float*)d_in[0];
  const float* ln_g  = (const float*)d_in[1];
  const float* ln_b  = (const float*)d_in[2];
  const float* fus_w = (const float*)d_in[3];
  const float* fus_b = (const float*)d_in[4];
  const float *in_w[2], *conv_w[2], *conv_b[2], *xproj_w[2], *dt_w[2],
              *dt_b[2], *A_log[2], *Dp[2], *out_w[2];
  for (int p = 0; p < 2; ++p) {
    int o = 5 + 9 * p;
    in_w[p]    = (const float*)d_in[o + 0];
    conv_w[p]  = (const float*)d_in[o + 1];
    conv_b[p]  = (const float*)d_in[o + 2];
    xproj_w[p] = (const float*)d_in[o + 3];
    dt_w[p]    = (const float*)d_in[o + 4];
    dt_b[p]    = (const float*)d_in[o + 5];
    A_log[p]   = (const float*)d_in[o + 6];
    Dp[p]      = (const float*)d_in[o + 7];
    out_w[p]   = (const float*)d_in[o + 8];
  }

  // ---- workspace carve-up ----
  char* wsp = (char*)d_ws;
  size_t off = 0;
  auto alloc = [&](size_t bytes) -> void* {
    void* p = wsp + off;
    off = (off + bytes + 255) & ~(size_t)255;
    return p;
  };
  f16*   xn_h   = (f16*)alloc((size_t)ROWS * D_MODEL * 2);
  f16*   xnr_h  = (f16*)alloc((size_t)ROWS * D_MODEL * 2);
  f16*   w_in16[2], *w_xp16[2], *w_dt16[2], *w_out16[2];
  for (int p = 0; p < 2; ++p) {
    w_in16[p]  = (f16*)alloc((size_t)2 * D_INNER * D_MODEL * 2);
    w_xp16[p]  = (f16*)alloc((size_t)XPROJ_N * D_INNER * 2);
    w_dt16[p]  = (f16*)alloc((size_t)D_INNER * DT_RANK * 2);
    w_out16[p] = (f16*)alloc((size_t)D_MODEL * D_INNER * 2);
  }
  f16*   fus16  = (f16*)alloc((size_t)D_MODEL * 2 * D_INNER);
  float* xz     = (float*)alloc((size_t)ROWS * 2 * D_INNER * 4);
  float* xc32   = (float*)alloc((size_t)ROWS * D_INNER * 4);
  f16*   xc16   = (f16*)alloc((size_t)ROWS * D_INNER * 2);
  float* dbl    = (float*)alloc((size_t)ROWS * XPROJ_N * 4);
  f16*   dt16a  = (f16*)alloc((size_t)ROWS * DT_RANK * 2);
  float* delta  = (float*)alloc((size_t)ROWS * D_INNER * 4);
  f16*   yg16   = (f16*)alloc((size_t)ROWS * D_INNER * 2);
  f16*   feat16 = (f16*)alloc((size_t)ROWS * 2 * D_MODEL * 2);

  // ---- weight conversion to f16 ----
  auto cvt = [&](const float* src, f16* dst, int n) {
    int blocks = (n + 255) / 256; if (blocks > 16384) blocks = 16384;
    k_f32_to_f16<<<blocks, 256, 0, stream>>>(src, dst, n);
  };
  for (int p = 0; p < 2; ++p) {
    cvt(in_w[p],    w_in16[p],  2 * D_INNER * D_MODEL);
    cvt(xproj_w[p], w_xp16[p],  XPROJ_N * D_INNER);
    cvt(dt_w[p],    w_dt16[p],  D_INNER * DT_RANK);
    cvt(out_w[p],   w_out16[p], D_MODEL * D_INNER);
  }
  cvt(fus_w, fus16, D_MODEL * 2 * D_INNER);

  // ---- LayerNorm (+ flipped copy) ----
  k_layernorm<<<ROWS, 256, 0, stream>>>(x, ln_g, ln_b, xn_h, xnr_h);

  // ---- two Mamba directions (activation scratch reused) ----
  for (int p = 0; p < 2; ++p) {
    const f16* Adir = (p == 0) ? xn_h : xnr_h;
    // in_proj: (4096,1024) x (4096,1024)^T -> xz f32
    gemm(stream, Adir, D_MODEL, w_in16[p], D_MODEL,
         ROWS, 2 * D_INNER, D_MODEL, xz, nullptr, 2 * D_INNER,
         nullptr, nullptr, 0, 0, 0);
    // depthwise causal conv + SiLU
    k_conv_silu<<<(ROWS * D_INNER + 255) / 256, 256, 0, stream>>>(
        xz, conv_w[p], conv_b[p], xc32, xc16);
    // x_proj: -> dbl (B,L,96) f32   (N=96 -> simple kernel)
    gemm(stream, xc16, D_INNER, w_xp16[p], D_INNER,
         ROWS, XPROJ_N, D_INNER, dbl, nullptr, XPROJ_N,
         nullptr, nullptr, 0, 0, 0);
    // pack dt columns to f16
    k_extract_dt<<<(ROWS * DT_RANK + 255) / 256, 256, 0, stream>>>(
        dbl, dt16a, ROWS * DT_RANK);
    // dt_proj + bias + softplus -> delta f32
    gemm(stream, dt16a, DT_RANK, w_dt16[p], DT_RANK,
         ROWS, D_INNER, DT_RANK, delta, nullptr, D_INNER,
         dt_b[p], nullptr, 1, 0, 0);
    // selective scan (fused skip + gate) -> yg16
    k_scan<<<B_SZ * (D_INNER / 256), 256, 0, stream>>>(
        delta, xc32, dbl, xz, A_log[p], Dp[p], yg16);
    // out_proj -> feat16 (backward direction: flip over L, cols 1024..2047)
    gemm(stream, yg16, D_INNER, w_out16[p], D_INNER,
         ROWS, D_MODEL, D_INNER, nullptr, feat16, 2 * D_MODEL,
         nullptr, nullptr, 2, p == 0 ? 0 : D_MODEL, p);
  }

  // ---- fusion GEMM + bias + residual -> d_out (f32) ----
  gemm(stream, feat16, 2 * D_MODEL, fus16, 2 * D_MODEL,
       ROWS, D_MODEL, 2 * D_MODEL, (float*)d_out, nullptr, D_MODEL,
       fus_b, x, 3, 0, 0);
}
